// GATLSTMCell_23184233464534
// MI455X (gfx1250) — matrix operand order repaired and checked by reading, not compiled, not run on hardware
//
#include <hip/hip_runtime.h>
#include <hip/hip_bf16.h>
#include <math.h>

#define N_NODES 50000
#define N_EDGES 800000
#define IN_CH   32
#define HID     64
#define CPH     32
#define ED      16

typedef __attribute__((ext_vector_type(16))) _Float16 v16h;
typedef __attribute__((ext_vector_type(8)))  float    v8f;

__device__ __forceinline__ void atomicMaxF(float* addr, float v) {
    // works for all finite floats given init = -inf (0xFF800000)
    if (v >= 0.f) atomicMax((int*)addr, __float_as_int(v));
    else          atomicMin((unsigned int*)addr, __float_as_uint(v));
}

// ---------------- K0: init smax/-inf, den/0, agg/0 ----------------
__global__ void k0_init(float* __restrict__ smax, float* __restrict__ den,
                        float* __restrict__ agg, int nsd, int nagg) {
    int i = blockIdx.x * blockDim.x + threadIdx.x;
    int stride = gridDim.x * blockDim.x;
    for (int j = i; j < nagg; j += stride) agg[j] = 0.f;
    for (int j = i; j < nsd; j += stride) {
        den[j] = 0.f;
        ((unsigned int*)smax)[j] = 0xFF800000u; // -inf
    }
}

// ---------------- K1: edge MLP  e = gelu(ea@W1+b1)@W2+b2 ----------------
__global__ void k1_edge_mlp(const float* __restrict__ ea,
                            const float* __restrict__ Wm1, const float* __restrict__ bm1,
                            const float* __restrict__ Wm2, const float* __restrict__ bm2,
                            float* __restrict__ eo) {
    __shared__ float sW1[ED * ED], sW2[ED * ED], sb1[ED], sb2[ED];
    for (int i = threadIdx.x; i < ED * ED; i += blockDim.x) { sW1[i] = Wm1[i]; sW2[i] = Wm2[i]; }
    if (threadIdx.x < ED) { sb1[threadIdx.x] = bm1[threadIdx.x]; sb2[threadIdx.x] = bm2[threadIdx.x]; }
    __syncthreads();
    int e = blockIdx.x * blockDim.x + threadIdx.x;
    if (e >= N_EDGES) return;
    float a[ED], h1[ED];
#pragma unroll
    for (int k = 0; k < ED; k++) a[k] = ea[(size_t)e * ED + k];
#pragma unroll
    for (int j = 0; j < ED; j++) {
        float s = sb1[j];
#pragma unroll
        for (int k = 0; k < ED; k++) s += a[k] * sW1[k * ED + j];
        h1[j] = 0.5f * s * (1.f + erff(s * 0.70710678118654752f)); // exact GELU
    }
#pragma unroll
    for (int j = 0; j < ED; j++) {
        float s = sb2[j];
#pragma unroll
        for (int k = 0; k < ED; k++) s += h1[k] * sW2[k * ED + j];
        eo[(size_t)e * ED + j] = s;
    }
}

// ---------------- K2: node transforms via WMMA ----------------
// T[node][sc][64], sc = side*8 + path*4 + gate   (side0=lin_l, side1=lin_r)
__global__ void k2_node_wmma(const float* __restrict__ x, const float* __restrict__ hp,
                             const float* __restrict__ Wlx, const float* __restrict__ blx,
                             const float* __restrict__ Wrx, const float* __restrict__ brx,
                             const float* __restrict__ Wlh, const float* __restrict__ blh,
                             const float* __restrict__ Wrh, const float* __restrict__ brh,
                             float* __restrict__ T) {
    const int nt    = blockIdx.x;      // node tile (16 rows), N = 3125*16 exactly
    const int sc    = blockIdx.y;      // 0..15
    const int side  = sc >> 3;
    const int combo = sc & 7;
    const int path  = combo >> 2;      // 0 = x-path (K=32), 1 = h-path (K=64)
    const int gate  = combo & 3;
    const int wave  = threadIdx.x >> 5;   // col tile 0..3
    const int lane  = threadIdx.x & 31;

    const int K = path ? HID : IN_CH;
    const float* in = path ? hp : x;
    const float* W  = side ? (path ? Wrh : Wrx) : (path ? Wlh : Wlx);
    const float* Bv = side ? (path ? brh : brx) : (path ? blh : blx);
    W  += (size_t)gate * K * HID;
    Bv += gate * HID;

    const int ln15 = lane & 15;
    const int row  = nt * 16 + ln15;          // A row
    const int col  = wave * 16 + ln15;        // B/C col
    const int kb   = (lane & 16) ? 8 : 0;     // A K-offset per ISA layout
    const int kbB  = (lane & 16) ? 16 : 0;    // unused; B is loaded with kb, same k order as A

    const float* arow = in + (size_t)row * K;
    v16h a0{}, a1{}, b0{}, b1{};
#pragma unroll
    for (int i = 0; i < 8; i++) {
        a0[i]     = (_Float16)arow[kb + i];
        a0[8 + i] = (_Float16)arow[16 + kb + i];
    }
#pragma unroll
    for (int i = 0; i < 16; i++) b0[i] = (_Float16)W[(size_t)(kb + i + (i & 8)) * HID + col];
    if (path) { // uniform branch per block, EXEC stays all-ones
#pragma unroll
        for (int i = 0; i < 8; i++) {
            a1[i]     = (_Float16)arow[32 + kb + i];
            a1[8 + i] = (_Float16)arow[48 + kb + i];
        }
#pragma unroll
        for (int i = 0; i < 16; i++) b1[i] = (_Float16)W[(size_t)(32 + kb + i + (i & 8)) * HID + col];
    }
    float bias = Bv[col];
    v8f c;
#pragma unroll
    for (int i = 0; i < 8; i++) c[i] = bias;
    c = __builtin_amdgcn_wmma_f32_16x16x32_f16(false, a0, false, b0, (short)0, c, false, false);
    if (path)
        c = __builtin_amdgcn_wmma_f32_16x16x32_f16(false, a1, false, b1, (short)0, c, false, false);
    const int mrow = (lane & 16) ? 8 : 0;
#pragma unroll
    for (int i = 0; i < 8; i++) {
        int node = nt * 16 + mrow + i;
        T[((size_t)node * 16 + sc) * HID + col] = c[i];
    }
}

// ---------------- K3: per-edge attention score + atomic max ----------------
__global__ void k3_score(const int* __restrict__ ei, const float* __restrict__ eM,
                         const float* __restrict__ Wex, const float* __restrict__ Weh,
                         const float* __restrict__ attx, const float* __restrict__ atth,
                         const float* __restrict__ T,
                         float* __restrict__ score, float* __restrict__ smax) {
    __shared__ float sWe[2 * 4 * ED * HID]; // 8192 floats
    __shared__ float sAtt[2 * 4 * HID];     // 512 floats
    for (int i = threadIdx.x; i < 4 * ED * HID; i += blockDim.x) {
        sWe[i] = Wex[i]; sWe[4 * ED * HID + i] = Weh[i];
    }
    for (int i = threadIdx.x; i < 4 * HID; i += blockDim.x) {
        sAtt[i] = attx[i]; sAtt[4 * HID + i] = atth[i];
    }
    __syncthreads();
    int e = blockIdx.x * blockDim.x + threadIdx.x;
    if (e >= N_EDGES) return;
    int src = ei[e], dst = ei[N_EDGES + e];
    float ev[ED];
#pragma unroll
    for (int k = 0; k < ED; k++) ev[k] = eM[(size_t)e * ED + k];
    for (int combo = 0; combo < 8; combo++) {
        const float* We = sWe + (combo >> 2) * (4 * ED * HID) + (combo & 3) * (ED * HID);
        const float* at = sAtt + (combo >> 2) * (4 * HID) + (combo & 3) * HID;
        const float* XL = T + ((size_t)src * 16 + combo) * HID;
        const float* XR = T + ((size_t)dst * 16 + 8 + combo) * HID;
        float acc0 = 0.f, acc1 = 0.f;
        for (int c = 0; c < HID; c++) {
            float s = XL[c] + XR[c];
#pragma unroll
            for (int k = 0; k < ED; k++) s += ev[k] * We[k * HID + c];
            float lr = s > 0.f ? s : 0.2f * s;
            float t = lr * at[c];
            if (c < CPH) acc0 += t; else acc1 += t;
        }
        score[(size_t)e * 16 + combo * 2 + 0] = acc0;
        score[(size_t)e * 16 + combo * 2 + 1] = acc1;
        atomicMaxF(smax + (size_t)dst * 16 + combo * 2 + 0, acc0);
        atomicMaxF(smax + (size_t)dst * 16 + combo * 2 + 1, acc1);
    }
}

// ---------------- K4: scatter exp + weighted aggregation ----------------
__global__ void k4_scatter(const int* __restrict__ ei,
                           const float* __restrict__ score, const float* __restrict__ smax,
                           const float* __restrict__ T,
                           float* __restrict__ den, float* __restrict__ agg) {
    int e = blockIdx.x * blockDim.x + threadIdx.x;
    if (e >= N_EDGES) return;
    int src = ei[e], dst = ei[N_EDGES + e];
    for (int combo = 0; combo < 8; combo++) {
        float ex0 = __expf(score[(size_t)e * 16 + combo * 2 + 0] - smax[(size_t)dst * 16 + combo * 2 + 0]);
        float ex1 = __expf(score[(size_t)e * 16 + combo * 2 + 1] - smax[(size_t)dst * 16 + combo * 2 + 1]);
        unsafeAtomicAdd(den + (size_t)dst * 16 + combo * 2 + 0, ex0);
        unsafeAtomicAdd(den + (size_t)dst * 16 + combo * 2 + 1, ex1);
        const float* XL = T + ((size_t)src * 16 + combo) * HID;
        float* A = agg + ((size_t)dst * 8 + combo) * HID;
#pragma unroll
        for (int c = 0; c < HID; c++)
            unsafeAtomicAdd(A + c, (c < CPH ? ex0 : ex1) * XL[c]);
    }
}

// ---------------- K5: normalize + LSTM + LayerNorm ----------------
__global__ void k5_final(const float* __restrict__ agg, const float* __restrict__ den,
                         const float* __restrict__ biasx, const float* __restrict__ biash,
                         const float* __restrict__ c_prev, const float* __restrict__ gamma,
                         const float* __restrict__ beta, float* __restrict__ out) {
    __shared__ float s1[HID], s2[HID];
    int n = blockIdx.x, c = threadIdx.x;
    int hd = c >> 5; // head
    float g4[4];
#pragma unroll
    for (int g = 0; g < 4; g++) {
        float ax = agg[((size_t)n * 8 + g) * HID + c] /
                   (den[(size_t)n * 16 + g * 2 + hd] + 1e-16f) + biasx[g * HID + c];
        float ah = agg[((size_t)n * 8 + 4 + g) * HID + c] /
                   (den[(size_t)n * 16 + (4 + g) * 2 + hd] + 1e-16f) + biash[g * HID + c];
        g4[g] = ax + ah;
    }
    float it = 1.f / (1.f + __expf(-g4[0]));
    float ft = 1.f / (1.f + __expf(-g4[1]));
    float ot = 1.f / (1.f + __expf(-g4[2]));
    float gt = tanhf(g4[3]);
    float ct = ft * c_prev[(size_t)n * HID + c] + it * gt;
    float ht = ot * tanhf(ct);
    s1[c] = ht; s2[c] = ht * ht;
    __syncthreads();
    for (int off = 32; off > 0; off >>= 1) {
        if (c < off) { s1[c] += s1[c + off]; s2[c] += s2[c + off]; }
        __syncthreads();
    }
    float mu  = s1[0] * (1.f / HID);
    float var = s2[0] * (1.f / HID) - mu * mu;
    float hn = (ht - mu) * rsqrtf(var + 1e-5f) * gamma[c] + beta[c];
    out[(size_t)n * HID + c] = hn;                             // h_t
    out[(size_t)N_NODES * HID + (size_t)n * HID + c] = ct;     // c_t
}

extern "C" void kernel_launch(void* const* d_in, const int* in_sizes, int n_in,
                              void* d_out, int out_size, void* d_ws, size_t ws_size,
                              hipStream_t stream) {
    const float* x_t    = (const float*)d_in[0];
    const float* h_prev = (const float*)d_in[1];
    const float* c_prev = (const float*)d_in[2];
    const int*   ei     = (const int*)d_in[3];
    const float* eattr  = (const float*)d_in[4];
    const float* Wm1    = (const float*)d_in[5];
    const float* bm1    = (const float*)d_in[6];
    const float* Wm2    = (const float*)d_in[7];
    const float* bm2    = (const float*)d_in[8];
    const float* Wlx    = (const float*)d_in[9];
    const float* blx    = (const float*)d_in[10];
    const float* Wrx    = (const float*)d_in[11];
    const float* brx    = (const float*)d_in[12];
    const float* Wex    = (const float*)d_in[13];
    const float* attx   = (const float*)d_in[14];
    const float* biasx  = (const float*)d_in[15];
    const float* Wlh    = (const float*)d_in[16];
    const float* blh    = (const float*)d_in[17];
    const float* Wrh    = (const float*)d_in[18];
    const float* brh    = (const float*)d_in[19];
    const float* Weh    = (const float*)d_in[20];
    const float* atth   = (const float*)d_in[21];
    const float* biash  = (const float*)d_in[22];
    const float* gamma  = (const float*)d_in[23];
    const float* beta   = (const float*)d_in[24];

    float* ws    = (float*)d_ws;
    float* T     = ws;                               // N*16*64
    float* eM    = T     + (size_t)N_NODES * 16 * 64; // E*16
    float* score = eM    + (size_t)N_EDGES * 16;      // E*16
    float* smax  = score + (size_t)N_EDGES * 16;      // N*16
    float* den   = smax  + (size_t)N_NODES * 16;      // N*16
    float* agg   = den   + (size_t)N_NODES * 16;      // N*8*64

    k0_init<<<100000, 256, 0, stream>>>(smax, den, agg, N_NODES * 16, N_NODES * 8 * 64);
    k1_edge_mlp<<<(N_EDGES + 255) / 256, 256, 0, stream>>>(eattr, Wm1, bm1, Wm2, bm2, eM);
    dim3 g2(N_NODES / 16, 16);
    k2_node_wmma<<<g2, 128, 0, stream>>>(x_t, h_prev, Wlx, blx, Wrx, brx,
                                         Wlh, blh, Wrh, brh, T);
    k3_score<<<(N_EDGES + 255) / 256, 256, 0, stream>>>(ei, eM, Wex, Weh, attx, atth,
                                                        T, score, smax);
    k4_scatter<<<(N_EDGES + 255) / 256, 256, 0, stream>>>(ei, score, smax, T, den, agg);
    k5_final<<<N_NODES, 64, 0, stream>>>(agg, den, biasx, biash, c_prev, gamma, beta,
                                         (float*)d_out);
}
